// EDGEAttention_4466765988283
// MI455X (gfx1250) — matrix operand order repaired and checked by reading, compiled, not run on hardware
//
#include <hip/hip_runtime.h>

// ---------------------------------------------------------------------------
// Problem constants (from reference)
// ---------------------------------------------------------------------------
#define BSZ   8
#define SEQ   1024
#define DMODEL 512
#define NHEAD 8
#define HDIM  64
#define MROWS (BSZ * SEQ)        // 8192
#define D3    (3 * DMODEL)       // 1536
#define LN_EPS 1e-5f

// ---------------------------------------------------------------------------
// CDNA5 WMMA / vector types
// ---------------------------------------------------------------------------
typedef __bf16        bf16_t;
typedef __attribute__((ext_vector_type(16))) __bf16 v16bf;
typedef __attribute__((ext_vector_type(8)))  float  v8f;
typedef __attribute__((ext_vector_type(4)))  unsigned int u32x4;
typedef __attribute__((ext_vector_type(2)))  unsigned int u32x2;

union Frag16 {            // 16 bf16 = 32 bytes = two 16B chunks
    v16bf v;
    u32x4 q[2];
};

__device__ __forceinline__ v8f wmma_bf16(v16bf a, v16bf b, v8f c) {
    // D = A(16x32 bf16) x B(32x16 bf16) + C(16x16 f32)
    return __builtin_amdgcn_wmma_f32_16x16x32_bf16(
        /*neg_a=*/false, a, /*neg_b=*/false, b,
        /*c_mod=*/(short)0, c, /*reuse_a=*/false, /*reuse_b=*/false);
}

__device__ __forceinline__ v8f v8f_zero() {
    v8f z = {0.f, 0.f, 0.f, 0.f, 0.f, 0.f, 0.f, 0.f};
    return z;
}

// CDNA5 async global -> LDS copy (ASYNCcnt-tracked), 16 bytes per lane.
__device__ __forceinline__ void async_load_b128(unsigned lds_addr, const void* gaddr) {
    asm volatile("global_load_async_to_lds_b128 %0, %1, off"
                 :: "v"(lds_addr), "v"(gaddr)
                 : "memory");
}
__device__ __forceinline__ void wait_asynccnt0() {
    asm volatile("s_wait_asynccnt 0x0" ::: "memory");
}
// Wave-local LDS ordering (store -> cross-lane load within one wave).
__device__ __forceinline__ void wait_dscnt0() {
    asm volatile("s_wait_dscnt 0x0" ::: "memory");
}

// ---------------------------------------------------------------------------
// Elementwise converts
// ---------------------------------------------------------------------------
__global__ void cvt_f32_to_bf16(const float* __restrict__ src,
                                bf16_t* __restrict__ dst, int n) {
    int i = blockIdx.x * 256 + threadIdx.x;
    if (i < n) dst[i] = (bf16_t)src[i];
}

// src: (K, N) f32 row-major  ->  dst: (N, K) bf16 row-major (i.e. B^T, K contiguous)
__global__ void transpose_to_bf16(const float* __restrict__ src,
                                  bf16_t* __restrict__ dst, int K, int N) {
    int i = blockIdx.x * 256 + threadIdx.x;
    if (i < K * N) {
        int n = i / K;
        int k = i - n * K;
        dst[i] = (bf16_t)src[(size_t)k * N + n];
    }
}

// Pack bool mask (1 byte/elem, values 0/1) into bitmask words: 32 keys / uint32.
__global__ void pack_mask_kernel(const unsigned char* __restrict__ m,
                                 unsigned* __restrict__ w, int nWords) {
    int i = blockIdx.x * 256 + threadIdx.x;
    if (i >= nWords) return;
    u32x4 a = *(const u32x4*)(m + (size_t)i * 32);
    u32x4 b = *(const u32x4*)(m + (size_t)i * 32 + 16);
    unsigned r = 0;
    #pragma unroll
    for (int j = 0; j < 4; ++j) {
        unsigned u = a[j];
        r |= ((u & 1u) | ((u >> 7) & 2u) | ((u >> 14) & 4u) | ((u >> 21) & 8u)) << (4 * j);
    }
    #pragma unroll
    for (int j = 0; j < 4; ++j) {
        unsigned u = b[j];
        r |= ((u & 1u) | ((u >> 7) & 2u) | ((u >> 14) & 4u) | ((u >> 21) & 8u)) << (16 + 4 * j);
    }
    w[i] = r;
}

// ---------------------------------------------------------------------------
// Generic bf16 WMMA GEMM:  C(M,N) = A(M,K) @ Bt(N,K)^T + bias
//   EPI 0: store f32 row-major to Cf
//   EPI 1: relu, store bf16 row-major to Cb
//   EPI 2: QKV epilogue: scatter into Q (b,h,L,hd; scaled), K (b,h,L,hd), Vt (b,h,hd,L)
// Block: 256 threads = 8 waves; tile 128(M) x 64(N); K-step 32.
// Tiles are staged with double-buffered global_load_async_to_lds_b128.
// ---------------------------------------------------------------------------
template <int EPI>
__global__ __launch_bounds__(256, 2)
void gemm_bf16_kernel(const bf16_t* __restrict__ A,
                      const bf16_t* __restrict__ Bt,
                      const float* __restrict__ bias,
                      float* __restrict__ Cf,
                      bf16_t* __restrict__ Cb,
                      bf16_t* __restrict__ Qo,
                      bf16_t* __restrict__ Ko,
                      bf16_t* __restrict__ Vto,
                      int M, int N, int K, float qscale) {
    __shared__ bf16_t As[2][128 * 32];   // 2 x 8 KB
    __shared__ bf16_t Bs[2][64 * 32];    // 2 x 4 KB

    const int tid    = threadIdx.x;
    const int lane   = tid & 31;
    const int wave   = tid >> 5;         // 0..7
    const int waveM  = wave >> 1;        // 0..3
    const int waveN  = wave & 1;         // 0..1
    const int laneLo = lane & 15;
    const int laneHi = lane >> 4;

    const int blockRow = blockIdx.y * 128;
    const int blockCol = blockIdx.x * 64;

    // per-thread staging assignments (3 x 16B async copies per thread)
    const int ra0 = (tid * 2) >> 2,       ka0 = ((tid * 2) & 3) * 8;
    const int ra1 = (tid * 2 + 1) >> 2,   ka1 = ((tid * 2 + 1) & 3) * 8;
    const int nb0 = tid >> 2,             kb0 = (tid & 3) * 8;

    auto issue_tile = [&](int k0, int buf) {
        async_load_b128((unsigned)(size_t)(void*)&As[buf][ra0 * 32 + ka0],
                        &A[(size_t)(blockRow + ra0) * K + k0 + ka0]);
        async_load_b128((unsigned)(size_t)(void*)&As[buf][ra1 * 32 + ka1],
                        &A[(size_t)(blockRow + ra1) * K + k0 + ka1]);
        async_load_b128((unsigned)(size_t)(void*)&Bs[buf][nb0 * 32 + kb0],
                        &Bt[(size_t)(blockCol + nb0) * K + k0 + kb0]);
    };

    v8f acc[2][2];
    acc[0][0] = v8f_zero(); acc[0][1] = v8f_zero();
    acc[1][0] = v8f_zero(); acc[1][1] = v8f_zero();

    issue_tile(0, 0);
    const int nk = K >> 5;
    for (int ki = 0; ki < nk; ++ki) {
        const int buf = ki & 1;
        wait_asynccnt0();       // my async writes into buf are done
        __syncthreads();        // everyone's writes done; prev buf reads done
        if (ki + 1 < nk) issue_tile((ki + 1) << 5, buf ^ 1);

        Frag16 a[2], b[2];
        #pragma unroll
        for (int mt = 0; mt < 2; ++mt) {
            int r = waveM * 32 + mt * 16 + laneLo;
            a[mt].q[0] = *(const u32x4*)&As[buf][r * 32 + laneHi * 16];
            a[mt].q[1] = *(const u32x4*)&As[buf][r * 32 + laneHi * 16 + 8];
        }
        #pragma unroll
        for (int nt = 0; nt < 2; ++nt) {
            int n = waveN * 32 + nt * 16 + laneLo;
            b[nt].q[0] = *(const u32x4*)&Bs[buf][n * 32 + laneHi * 16];
            b[nt].q[1] = *(const u32x4*)&Bs[buf][n * 32 + laneHi * 16 + 8];
        }
        #pragma unroll
        for (int mt = 0; mt < 2; ++mt)
            #pragma unroll
            for (int nt = 0; nt < 2; ++nt)
                acc[mt][nt] = wmma_bf16(a[mt].v, b[nt].v, acc[mt][nt]);
    }

    // --- epilogue ---
    #pragma unroll
    for (int mt = 0; mt < 2; ++mt) {
        #pragma unroll
        for (int nt = 0; nt < 2; ++nt) {
            int col = blockCol + waveN * 32 + nt * 16 + laneLo;
            float bv = bias ? bias[col] : 0.f;
            #pragma unroll
            for (int v = 0; v < 8; ++v) {
                int   row = blockRow + waveM * 32 + mt * 16 + v + 8 * laneHi;
                float val = acc[mt][nt][v] + bv;
                if (EPI == 0) {
                    Cf[(size_t)row * N + col] = val;
                } else if (EPI == 1) {
                    Cb[(size_t)row * N + col] = (bf16_t)fmaxf(val, 0.f);
                } else { // EPI == 2: QKV scatter
                    int b   = row >> 10;          // row / SEQ
                    int l   = row & (SEQ - 1);
                    int c   = col;
                    if (c < DMODEL) {
                        int h = c >> 6, hd = c & 63;
                        Qo[(((size_t)(b * NHEAD + h)) * SEQ + l) * HDIM + hd] =
                            (bf16_t)(val * qscale);
                    } else if (c < 2 * DMODEL) {
                        c -= DMODEL;
                        int h = c >> 6, hd = c & 63;
                        Ko[(((size_t)(b * NHEAD + h)) * SEQ + l) * HDIM + hd] =
                            (bf16_t)val;
                    } else {
                        c -= 2 * DMODEL;
                        int h = c >> 6, hd = c & 63;
                        Vto[(((size_t)(b * NHEAD + h)) * HDIM + hd) * SEQ + l] =
                            (bf16_t)val;
                    }
                }
            }
        }
    }
}

// ---------------------------------------------------------------------------
// Flash attention. One workgroup = 4 waves = 64 query rows of one (b,h).
// Each wave owns 16 query rows; loops over keys in blocks of 64.
//   S = Q@K^T (scale folded into Q), bitmask, online softmax, O += P@V.
// Q: (b,h,L,hd) bf16 ; K: (b,h,L,hd) bf16 ; Vt: (b,h,hd,L) bf16
// maskW: packed bitmask, 32 keys per uint32, (b, L, L/32)
// Out: (b, L, d) f32 with head-concat columns.
// ---------------------------------------------------------------------------
__global__ __launch_bounds__(128, 2)
void flash_attn_kernel(const bf16_t* __restrict__ Q,
                       const bf16_t* __restrict__ K,
                       const bf16_t* __restrict__ Vt,
                       const unsigned* __restrict__ maskW,
                       float* __restrict__ Out) {
    __shared__ bf16_t Plds[4][16][64];   // per-wave P staging, 8 KB

    const int lane   = threadIdx.x & 31;
    const int wave   = threadIdx.x >> 5;   // 0..3
    const int laneLo = lane & 15;
    const int laneHi = lane >> 4;

    const int bh    = blockIdx.y;          // 0..63
    const int b     = bh >> 3;
    const int h     = bh & 7;
    const int qbase = blockIdx.x * 64 + wave * 16;

    const bf16_t* Qbh  = Q  + (size_t)bh * SEQ * HDIM;
    const bf16_t* Kbh  = K  + (size_t)bh * SEQ * HDIM;
    const bf16_t* Vtbh = Vt + (size_t)bh * HDIM * SEQ;

    // Preload Q A-fragments (16 rows x hd=64 -> 2 frags of K=32)
    Frag16 qa[2];
    #pragma unroll
    for (int kt = 0; kt < 2; ++kt) {
        const bf16_t* qp = Qbh + (size_t)(qbase + laneLo) * HDIM + kt * 32;
        qa[kt].q[0] = *(const u32x4*)(qp + laneHi * 16);
        qa[kt].q[1] = *(const u32x4*)(qp + laneHi * 16 + 8);
    }

    float runM[8], runL[8];
    #pragma unroll
    for (int v = 0; v < 8; ++v) { runM[v] = -1e30f; runL[v] = 0.f; }
    v8f o[4];
    #pragma unroll
    for (int t = 0; t < 4; ++t) o[t] = v8f_zero();

    for (int kb = 0; kb < SEQ; kb += 64) {
        // ---- S tile: 16 (q) x 64 (keys), GEMM-K = hd = 64 ----
        v8f s[4];
        #pragma unroll
        for (int nt = 0; nt < 4; ++nt) {
            const bf16_t* kp = Kbh + (size_t)(kb + nt * 16 + laneLo) * HDIM;
            Frag16 bk0, bk1;
            bk0.q[0] = *(const u32x4*)(kp + laneHi * 16);
            bk0.q[1] = *(const u32x4*)(kp + laneHi * 16 + 8);
            bk1.q[0] = *(const u32x4*)(kp + 32 + laneHi * 16);
            bk1.q[1] = *(const u32x4*)(kp + 32 + laneHi * 16 + 8);
            v8f sv = v8f_zero();
            sv = wmma_bf16(qa[0].v, bk0.v, sv);
            sv = wmma_bf16(qa[1].v, bk1.v, sv);
            s[nt] = sv;
        }

        // ---- bitmask + online softmax (per-lane rows v + 8*laneHi) ----
        #pragma unroll
        for (int v = 0; v < 8; ++v) {
            int qrow = qbase + v + 8 * laneHi;
            u32x2 mw = *(const u32x2*)&maskW[((size_t)b * SEQ + qrow) * (SEQ / 32)
                                             + (kb >> 5)];
            float x0 = ((mw.x >> laneLo)        & 1u) ? -1e30f : s[0][v];
            float x1 = ((mw.x >> (16 + laneLo)) & 1u) ? -1e30f : s[1][v];
            float x2 = ((mw.y >> laneLo)        & 1u) ? -1e30f : s[2][v];
            float x3 = ((mw.y >> (16 + laneLo)) & 1u) ? -1e30f : s[3][v];

            float rmax = fmaxf(fmaxf(x0, x1), fmaxf(x2, x3));
            #pragma unroll
            for (int off = 1; off < 16; off <<= 1)
                rmax = fmaxf(rmax, __shfl_xor(rmax, off, 32));

            float mnew  = fmaxf(runM[v], rmax);
            float alpha = __expf(runM[v] - mnew);
            x0 = __expf(x0 - mnew);
            x1 = __expf(x1 - mnew);
            x2 = __expf(x2 - mnew);
            x3 = __expf(x3 - mnew);

            float rs = (x0 + x1) + (x2 + x3);
            #pragma unroll
            for (int off = 1; off < 16; off <<= 1)
                rs += __shfl_xor(rs, off, 32);

            runL[v] = runL[v] * alpha + rs;
            runM[v] = mnew;
            o[0][v] *= alpha; o[1][v] *= alpha;
            o[2][v] *= alpha; o[3][v] *= alpha;

            Plds[wave][v + 8 * laneHi][laneLo]      = (bf16_t)x0;
            Plds[wave][v + 8 * laneHi][16 + laneLo] = (bf16_t)x1;
            Plds[wave][v + 8 * laneHi][32 + laneLo] = (bf16_t)x2;
            Plds[wave][v + 8 * laneHi][48 + laneLo] = (bf16_t)x3;
        }
        wait_dscnt0();   // wave-private tile: order P stores before A-layout reads

        // ---- reload P as A-fragments (16x64 -> 2 frags), O += P @ Vblock ----
        Frag16 pa[2];
        #pragma unroll
        for (int kt = 0; kt < 2; ++kt) {
            pa[kt].q[0] = *(const u32x4*)&Plds[wave][laneLo][kt * 32 + laneHi * 16];
            pa[kt].q[1] = *(const u32x4*)&Plds[wave][laneLo][kt * 32 + laneHi * 16 + 8];
        }
        #pragma unroll
        for (int t = 0; t < 4; ++t) {
            const bf16_t* vp = Vtbh + (size_t)(t * 16 + laneLo) * SEQ + kb;
            Frag16 bv0, bv1;
            bv0.q[0] = *(const u32x4*)(vp + laneHi * 16);
            bv0.q[1] = *(const u32x4*)(vp + laneHi * 16 + 8);
            bv1.q[0] = *(const u32x4*)(vp + 32 + laneHi * 16);
            bv1.q[1] = *(const u32x4*)(vp + 32 + laneHi * 16 + 8);
            o[t] = wmma_bf16(pa[0].v, bv0.v, o[t]);
            o[t] = wmma_bf16(pa[1].v, bv1.v, o[t]);
        }
    }

    // ---- normalize and write (b, L, d) with head-concat columns ----
    #pragma unroll
    for (int v = 0; v < 8; ++v) runL[v] = 1.0f / runL[v];
    #pragma unroll
    for (int t = 0; t < 4; ++t) {
        #pragma unroll
        for (int v = 0; v < 8; ++v) {
            int row = qbase + v + 8 * laneHi;
            int col = h * HDIM + t * 16 + laneLo;
            Out[((size_t)b * SEQ + row) * DMODEL + col] = o[t][v] * runL[v];
        }
    }
}

// ---------------------------------------------------------------------------
// Residual-add + LayerNorm over rows of length 512.
// One block per row, 256 threads, 2 elements each. Optional bf16 side copy.
// ---------------------------------------------------------------------------
__global__ __launch_bounds__(256, 4)
void add_ln_kernel(const float* __restrict__ X, const float* __restrict__ Y,
                   const float* __restrict__ gamma, const float* __restrict__ beta,
                   float* __restrict__ Out, bf16_t* __restrict__ Outb) {
    __shared__ float sm[2][8];
    const int t = threadIdx.x;
    const size_t base = (size_t)blockIdx.x * DMODEL;

    float v0 = X[base + t]       + Y[base + t];
    float v1 = X[base + t + 256] + Y[base + t + 256];
    float s  = v0 + v1;
    float ss = v0 * v0 + v1 * v1;
    #pragma unroll
    for (int off = 16; off > 0; off >>= 1) {
        s  += __shfl_xor(s,  off, 32);
        ss += __shfl_xor(ss, off, 32);
    }
    if ((t & 31) == 0) { sm[0][t >> 5] = s; sm[1][t >> 5] = ss; }
    __syncthreads();
    float tot = 0.f, tot2 = 0.f;
    #pragma unroll
    for (int i = 0; i < 8; ++i) { tot += sm[0][i]; tot2 += sm[1][i]; }

    const float mean = tot * (1.0f / DMODEL);
    const float var  = tot2 * (1.0f / DMODEL) - mean * mean;
    const float rstd = rsqrtf(var + LN_EPS);

    float o0 = (v0 - mean) * rstd * gamma[t]       + beta[t];
    float o1 = (v1 - mean) * rstd * gamma[t + 256] + beta[t + 256];
    Out[base + t]       = o0;
    Out[base + t + 256] = o1;
    if (Outb) {
        Outb[base + t]       = (bf16_t)o0;
        Outb[base + t + 256] = (bf16_t)o1;
    }
}

// ---------------------------------------------------------------------------
// Host-side orchestration
// ---------------------------------------------------------------------------
extern "C" void kernel_launch(void* const* d_in, const int* in_sizes, int n_in,
                              void* d_out, int out_size, void* d_ws, size_t ws_size,
                              hipStream_t stream) {
    (void)in_sizes; (void)n_in; (void)out_size; (void)ws_size;

    const float*         edge_x = (const float*)d_in[0];
    const unsigned char* emask  = (const unsigned char*)d_in[1];   // bool mask
    const float*         Wqkv   = (const float*)d_in[2];           // (512, 1536)
    const float*         bqkv   = (const float*)d_in[3];
    const float*         W1     = (const float*)d_in[4];
    const float*         b1     = (const float*)d_in[5];
    const float*         W2     = (const float*)d_in[6];
    const float*         b2     = (const float*)d_in[7];
    const float*         g1     = (const float*)d_in[8];
    const float*         be1    = (const float*)d_in[9];
    const float*         g2     = (const float*)d_in[10];
    const float*         be2    = (const float*)d_in[11];
    float*               out    = (float*)d_out;

    // workspace carve-up (256 B aligned)
    char* w = (char*)d_ws;
    auto carve = [&](size_t bytes) {
        void* p = (void*)w;
        w += (bytes + 255) & ~(size_t)255;
        return p;
    };
    bf16_t*   Xb    = (bf16_t*)carve((size_t)MROWS * DMODEL * 2);
    bf16_t*   Wqkvt = (bf16_t*)carve((size_t)D3 * DMODEL * 2);
    bf16_t*   W1t   = (bf16_t*)carve((size_t)DMODEL * DMODEL * 2);
    bf16_t*   W2t   = (bf16_t*)carve((size_t)DMODEL * DMODEL * 2);
    bf16_t*   Qs    = (bf16_t*)carve((size_t)MROWS * DMODEL * 2);
    bf16_t*   Ks    = (bf16_t*)carve((size_t)MROWS * DMODEL * 2);
    bf16_t*   Vts   = (bf16_t*)carve((size_t)MROWS * DMODEL * 2);
    float*    attnO = (float*)carve((size_t)MROWS * DMODEL * 4);
    float*    X1    = (float*)carve((size_t)MROWS * DMODEL * 4);
    bf16_t*   X1b   = (bf16_t*)carve((size_t)MROWS * DMODEL * 2);
    bf16_t*   Hb    = (bf16_t*)carve((size_t)MROWS * DMODEL * 2);
    float*    Ff    = (float*)carve((size_t)MROWS * DMODEL * 4);
    unsigned* maskW = (unsigned*)carve((size_t)BSZ * SEQ * (SEQ / 32) * 4);

    const float qscale = 0.125f;  // 1/sqrt(64)

    // 1) convert X, transpose weights to bf16 (N x K), pack mask to bits
    {
        int n = MROWS * DMODEL;
        cvt_f32_to_bf16<<<(n + 255) / 256, 256, 0, stream>>>(edge_x, Xb, n);
    }
    {
        int n = DMODEL * D3;
        transpose_to_bf16<<<(n + 255) / 256, 256, 0, stream>>>(Wqkv, Wqkvt, DMODEL, D3);
    }
    {
        int n = DMODEL * DMODEL;
        transpose_to_bf16<<<(n + 255) / 256, 256, 0, stream>>>(W1, W1t, DMODEL, DMODEL);
        transpose_to_bf16<<<(n + 255) / 256, 256, 0, stream>>>(W2, W2t, DMODEL, DMODEL);
    }
    {
        int nw = BSZ * SEQ * (SEQ / 32);
        pack_mask_kernel<<<(nw + 255) / 256, 256, 0, stream>>>(emask, maskW, nw);
    }

    // 2) fused QKV projection GEMM with scatter epilogue
    gemm_bf16_kernel<2><<<dim3(D3 / 64, MROWS / 128), 256, 0, stream>>>(
        Xb, Wqkvt, bqkv, nullptr, nullptr, Qs, Ks, Vts,
        MROWS, D3, DMODEL, qscale);

    // 3) flash attention: grid = (L/64, b*h)
    flash_attn_kernel<<<dim3(SEQ / 64, BSZ * NHEAD), 128, 0, stream>>>(
        Qs, Ks, Vts, maskW, attnO);

    // 4) x1 = LN(edge_x + attn) ; also bf16 copy for FFN
    add_ln_kernel<<<MROWS, 256, 0, stream>>>(edge_x, attnO, g1, be1, X1, X1b);

    // 5) H = relu(x1 @ W1 + b1)  (bf16 out)
    gemm_bf16_kernel<1><<<dim3(DMODEL / 64, MROWS / 128), 256, 0, stream>>>(
        X1b, W1t, b1, nullptr, Hb, nullptr, nullptr, nullptr,
        MROWS, DMODEL, DMODEL, 0.f);

    // 6) F = H @ W2 + b2  (f32 out)
    gemm_bf16_kernel<0><<<dim3(DMODEL / 64, MROWS / 128), 256, 0, stream>>>(
        Hb, W2t, b2, Ff, nullptr, nullptr, nullptr, nullptr,
        MROWS, DMODEL, DMODEL, 0.f);

    // 7) out = LN(x1 + F)
    add_ln_kernel<<<MROWS, 256, 0, stream>>>(X1, Ff, g2, be2, out, nullptr);
}
